// EncoderLayer_13941463843471
// MI455X (gfx1250) — compile-verified, matrix operand order checked by reading, not run on hardware
//
#include <hip/hip_runtime.h>
#include <hip/hip_bf16.h>
#include <math.h>

typedef _Float16 h16;
typedef h16  v16h __attribute__((ext_vector_type(16)));
typedef h16  h8   __attribute__((ext_vector_type(8)));
typedef float v8f __attribute__((ext_vector_type(8)));

#define DMODEL 128
#define DINNER 256
#define DSTATE 32
#define DTRANK 8
#define XDBLN  (DTRANK + 2*DSTATE)   // 72

// ---------------- f32 -> f16 conversion with zero tail padding ----------------
// out has `total` elements; elements >= nvalid are zero (pads weight rows to mult-of-32).
__global__ void cvt_pad_f16_kernel(const float* __restrict__ in, h16* __restrict__ out,
                                   int nvalid, int total) {
    int i = blockIdx.x * blockDim.x + threadIdx.x;
    if (i < total) out[i] = (i < nvalid) ? (h16)in[i] : (h16)0.f;
}

// ---------------- LayerNorm over last dim = 128 ----------------
__global__ void ln128_kernel(const float* __restrict__ x, const float* __restrict__ g,
                             const float* __restrict__ b, float* o32, h16* o16) {
    int row = blockIdx.x;
    int t   = threadIdx.x;                 // 128 threads
    __shared__ float sm[128];
    float v = x[(size_t)row * DMODEL + t];
    sm[t] = v; __syncthreads();
    for (int s = 64; s > 0; s >>= 1) { if (t < s) sm[t] += sm[t + s]; __syncthreads(); }
    float mu = sm[0] * (1.0f / 128.0f);
    __syncthreads();
    float d = v - mu;
    sm[t] = d * d; __syncthreads();
    for (int s = 64; s > 0; s >>= 1) { if (t < s) sm[t] += sm[t + s]; __syncthreads(); }
    float var = sm[0] * (1.0f / 128.0f);
    float y = d * rsqrtf(var + 1e-5f) * g[t] + b[t];
    if (o32) o32[(size_t)row * DMODEL + t] = y;
    if (o16) o16[(size_t)row * DMODEL + t] = (h16)y;
}

// ---------------- WMMA fragment load: two contiguous b128 loads per lane ----------------
// ISA 7.12.2 16-bit A/B layout per lane: halves [hi*8 .. hi*8+7] and [16+hi*8 .. 16+hi*8+7].
__device__ __forceinline__ v16h load_frag(const h16* __restrict__ p) {
    h8 lo = *(const h8*)p;
    h8 hi = *(const h8*)(p + 16);
    return __builtin_shufflevector(lo, hi, 0,1,2,3,4,5,6,7,8,9,10,11,12,13,14,15);
}

// ---------------- WMMA GEMM: C(M,N) = A(M,K) @ W(Npad,K)^T ----------------
// W zero-padded to Npad = mult of 32 rows -> inner loop is branch-free, EXEC all-ones.
// Each wave: one 16-row M tile x two 16-col N tiles (A fragment reused by 2 wmmas).
// Epilogue: +bias[n], +resid[m,n], act==1 -> exact GELU; optional f32/f16 stores.
template<int K>
__global__ void wmma_gemm_kernel(const h16* __restrict__ A, const h16* __restrict__ W,
                                 float* __restrict__ C32, h16* __restrict__ C16,
                                 const float* __restrict__ bias, const float* __restrict__ resid,
                                 int N, int act) {
    const int lane  = threadIdx.x;                        // 0..31 (wave32)
    const int mtile = blockIdx.y * blockDim.y + threadIdx.y;
    const int nt2   = blockIdx.x;                         // covers n-tiles 2*nt2, 2*nt2+1
    const int mlo   = lane & 15;
    const int hi    = lane >> 4;

    const h16* Arow  = A + (size_t)(mtile * 16 + mlo) * K + hi * 8;
    const h16* Wrow0 = W + (size_t)(nt2 * 32 + mlo) * K + hi * 8;
    const h16* Wrow1 = Wrow0 + (size_t)16 * K;

    v8f c0 = {0.f,0.f,0.f,0.f,0.f,0.f,0.f,0.f};
    v8f c1 = {0.f,0.f,0.f,0.f,0.f,0.f,0.f,0.f};
    #pragma unroll
    for (int k0 = 0; k0 < K; k0 += 32) {
        v16h a  = load_frag(Arow + k0);
        v16h b0 = load_frag(Wrow0 + k0);
        v16h b1 = load_frag(Wrow1 + k0);
        c0 = __builtin_amdgcn_wmma_f32_16x16x32_f16(false, a, false, b0, (short)0, c0, false, false);
        c1 = __builtin_amdgcn_wmma_f32_16x16x32_f16(false, a, false, b1, (short)0, c1, false, false);
    }

    #pragma unroll
    for (int half = 0; half < 2; ++half) {
        int  ncol = nt2 * 32 + half * 16 + mlo;
        v8f  c    = half ? c1 : c0;
        if (ncol < N) {
            float bv = bias ? bias[ncol] : 0.f;
            #pragma unroll
            for (int r = 0; r < 8; ++r) {
                int m = mtile * 16 + r + hi * 8;          // C layout: VGPR r -> M=r / r+8
                float v = c[r] + bv;
                size_t idx = (size_t)m * N + ncol;
                if (resid) v += resid[idx];
                if (act == 1) v = 0.5f * v * (1.0f + erff(v * 0.70710678f));
                if (C32) C32[idx] = v;
                if (C16) C16[idx] = (h16)v;
            }
        }
    }
}

// ---------------- depthwise causal conv + SiLU (direction-aware) ----------------
// xz: (B*L, 512), channels 0..255 = xc_raw. Output in direction-time order.
__global__ void conv_silu_kernel(const float* __restrict__ xz, const float* __restrict__ cw,
                                 const float* __restrict__ cb, float* __restrict__ xc32,
                                 h16* __restrict__ xc16, int L, int total, int dconv, int dir) {
    int id = blockIdx.x * blockDim.x + threadIdx.x;
    if (id >= total) return;
    int d  = id % DINNER;
    int bt = id / DINNER;
    int t  = bt % L;
    int b  = bt / L;
    float acc = 0.f;
    for (int k = 0; k < dconv; ++k) {
        int t2 = t - (dconv - 1) + k;
        if (t2 >= 0) {
            int src = dir ? (L - 1 - t2) : t2;
            acc += cw[d * dconv + k] * xz[(size_t)(b * L + src) * (2 * DINNER) + d];
        }
    }
    acc += cb[d];
    float s = acc / (1.f + __expf(-acc));             // SiLU
    size_t o = (size_t)(b * L + t) * DINNER + d;
    xc32[o] = s;
    xc16[o] = (h16)s;
}

// ---------------- selective scan (fused dt-proj + softplus + gating) ----------------
// One block per batch b; thread d owns channel d with h[32] state in VGPRs.
__global__ void scan_kernel(const float* __restrict__ xdbl, const float* __restrict__ dtw,
                            const float* __restrict__ dtb, const float* __restrict__ xc32,
                            const float* __restrict__ xz, const float* __restrict__ A_log,
                            const float* __restrict__ Dp, h16* __restrict__ yg16,
                            int L, int dir) {
    int b = blockIdx.x;
    int d = threadIdx.x;                               // 256 threads
    __shared__ float row[XDBLN];                       // dt_in(8) | B(32) | C(32)
    float A[DSTATE], h[DSTATE];
    #pragma unroll
    for (int s = 0; s < DSTATE; ++s) { A[s] = -__expf(A_log[d * DSTATE + s]); h[s] = 0.f; }
    float w[DTRANK];
    #pragma unroll
    for (int j = 0; j < DTRANK; ++j) w[j] = dtw[d * DTRANK + j];
    const float bdt = dtb[d], Dd = Dp[d];

    for (int t = 0; t < L; ++t) {
        __syncthreads();
        if (d < XDBLN) row[d] = xdbl[(size_t)(b * L + t) * XDBLN + d];
        __syncthreads();
        float dtin = bdt;
        #pragma unroll
        for (int j = 0; j < DTRANK; ++j) dtin += w[j] * row[j];
        float dt = (dtin > 20.f) ? dtin : log1pf(__expf(dtin));   // softplus
        float xt = xc32[(size_t)(b * L + t) * DINNER + d];
        float y = 0.f;
        #pragma unroll
        for (int s = 0; s < DSTATE; ++s) {
            float dA = __expf(dt * A[s]);
            h[s] = dA * h[s] + dt * row[DTRANK + s] * xt;
            y += h[s] * row[DTRANK + DSTATE + s];
        }
        int orig = dir ? (L - 1 - t) : t;
        float z = xz[(size_t)(b * L + orig) * (2 * DINNER) + DINNER + d];
        float g = (y + xt * Dd) * (z / (1.f + __expf(-z)));       // * silu(z)
        yg16[(size_t)(b * L + orig) * DINNER + d] = (h16)g;       // un-flipped position
    }
}

// ---------------- build concat([ff, fr]) in f16 ----------------
__global__ void cat_kernel(const float* __restrict__ f0, const float* __restrict__ f1,
                           h16* __restrict__ cat16, int total) {
    int id = blockIdx.x * blockDim.x + threadIdx.x;
    if (id >= total) return;
    int j = id % DMODEL, r = id / DMODEL;
    cat16[(size_t)r * (2 * DMODEL) + j]          = (h16)f0[id];
    cat16[(size_t)r * (2 * DMODEL) + DMODEL + j] = (h16)f1[id];
}

// ---------------- gated fuse: sigmoid(logit)*ff + (1-s)*fr -> f16 ----------------
__global__ void fuse_kernel(const float* __restrict__ logits, const float* __restrict__ ff,
                            const float* __restrict__ fr, h16* __restrict__ fused16, int n) {
    int id = blockIdx.x * blockDim.x + threadIdx.x;
    if (id >= n) return;
    float g = 1.f / (1.f + __expf(-logits[id]));
    fused16[id] = (h16)(g * ff[id] + (1.f - g) * fr[id]);
}

extern "C" void kernel_launch(void* const* d_in, const int* in_sizes, int n_in,
                              void* d_out, int out_size, void* d_ws, size_t ws_size,
                              hipStream_t stream) {
    const float* x    = (const float*)d_in[0];
    const float* n1g  = (const float*)d_in[1];
    const float* n1b  = (const float*)d_in[2];
    const float* big  = (const float*)d_in[3];
    const float* bib  = (const float*)d_in[4];
    const float* in_w[2]   = {(const float*)d_in[5],  (const float*)d_in[14]};
    const float* conv_w[2] = {(const float*)d_in[6],  (const float*)d_in[15]};
    const float* conv_b[2] = {(const float*)d_in[7],  (const float*)d_in[16]};
    const float* xp_w[2]   = {(const float*)d_in[8],  (const float*)d_in[17]};
    const float* dt_w[2]   = {(const float*)d_in[9],  (const float*)d_in[18]};
    const float* dt_b[2]   = {(const float*)d_in[10], (const float*)d_in[19]};
    const float* A_log[2]  = {(const float*)d_in[11], (const float*)d_in[20]};
    const float* Dp[2]     = {(const float*)d_in[12], (const float*)d_in[21]};
    const float* op_w[2]   = {(const float*)d_in[13], (const float*)d_in[22]};
    const float* gate_w = (const float*)d_in[23];
    const float* gate_b = (const float*)d_in[24];
    const float* out_w  = (const float*)d_in[25];
    const float* out_b  = (const float*)d_in[26];
    const float* fng    = (const float*)d_in[27];
    const float* fnb    = (const float*)d_in[28];
    const float* fc1w   = (const float*)d_in[29];
    const float* fc1b   = (const float*)d_in[30];
    const float* fc2w   = (const float*)d_in[31];
    const float* fc2b   = (const float*)d_in[32];
    const int dconv[2] = {4, 8};

    const int BL = in_sizes[0] / DMODEL;               // 65536
    const int L  = 2048;
    const int B  = BL / L;

    char* ws = (char*)d_ws;
    size_t off = 0;
    auto alloc = [&](size_t bytes) { size_t o = off; off += (bytes + 255) & ~(size_t)255; return o; };
    auto npad32 = [](int n) { return (n + 31) & ~31; };

    size_t o_xz   = alloc((size_t)BL * 512 * 4);       // xz; later x2 (base) + cat16 (+half)
    size_t o_xc32 = alloc((size_t)BL * 256 * 4);
    size_t o_xc16 = alloc((size_t)BL * 256 * 2);       // later h1_f16
    size_t o_xdbl = alloc((size_t)BL * XDBLN * 4);
    size_t o_yg16 = alloc((size_t)BL * 256 * 2);       // later ffn-LN f16 out
    size_t o_f0   = alloc((size_t)BL * 128 * 4);       // ff
    size_t o_f1   = alloc((size_t)BL * 128 * 4);       // fr
    size_t o_tmp  = alloc((size_t)BL * 128 * 4);       // LN1 tmp; later gate logits
    size_t o_xn16 = alloc((size_t)BL * 128 * 2);       // LN2 f16; later fused16
    size_t o_winw[2]  = {alloc(512 * 128 * 2),        alloc(512 * 128 * 2)};
    size_t o_wxpw[2]  = {alloc(96 * 256 * 2),         alloc(96 * 256 * 2)};   // 72 -> 96 pad
    size_t o_wopw[2]  = {alloc(128 * 256 * 2),        alloc(128 * 256 * 2)};
    size_t o_wgate = alloc(128 * 256 * 2);
    size_t o_woutw = alloc(128 * 128 * 2);
    size_t o_wfc1  = alloc(256 * 128 * 2);
    size_t o_wfc2  = alloc(128 * 256 * 2);
    size_t o_x2   = o_xz;                              // overlay (xz dead by then)
    size_t o_cat  = o_xz + ((size_t)BL * 512 * 4) / 2; // overlay at +67MB

    auto cvt = [&](const float* src, size_t o, int N, int K) {
        int total = npad32(N) * K;
        cvt_pad_f16_kernel<<<(total + 255) / 256, 256, 0, stream>>>(
            src, (h16*)(ws + o), N * K, total);
    };
    auto gemm = [&](size_t oA, size_t oW, float* C32, h16* C16, const float* bias,
                    const float* resid, int N, int K, int act) {
        dim3 blk(32, 4), grd(npad32(N) / 32, (BL / 16) / 4);
        if (K == 128)
            wmma_gemm_kernel<128><<<grd, blk, 0, stream>>>((const h16*)(ws + oA),
                (const h16*)(ws + oW), C32, C16, bias, resid, N, act);
        else
            wmma_gemm_kernel<256><<<grd, blk, 0, stream>>>((const h16*)(ws + oA),
                (const h16*)(ws + oW), C32, C16, bias, resid, N, act);
    };

    // ---- convert all GEMM weights to f16 (zero-padded rows) once ----
    for (int p = 0; p < 2; ++p) {
        cvt(in_w[p], o_winw[p], 512, 128);
        cvt(xp_w[p], o_wxpw[p], XDBLN, 256);
        cvt(op_w[p], o_wopw[p], 128, 256);
    }
    cvt(gate_w, o_wgate, 128, 256);
    cvt(out_w,  o_woutw, 128, 128);
    cvt(fc1w,   o_wfc1,  256, 128);
    cvt(fc2w,   o_wfc2,  128, 256);

    // ---- xn = LN(LN(x, norm1), bi.norm) -> f16 ----
    ln128_kernel<<<BL, 128, 0, stream>>>(x, n1g, n1b, (float*)(ws + o_tmp), nullptr);
    ln128_kernel<<<BL, 128, 0, stream>>>((const float*)(ws + o_tmp), big, bib,
                                         nullptr, (h16*)(ws + o_xn16));

    // ---- four mamba passes (2 param sets x 2 directions; in_proj shared per set) ----
    const int convTotal = B * L * DINNER;
    for (int p = 0; p < 2; ++p) {
        gemm(o_xn16, o_winw[p], (float*)(ws + o_xz), nullptr, nullptr, nullptr, 512, 128, 0);
        for (int dir = 0; dir < 2; ++dir) {
            conv_silu_kernel<<<(convTotal + 255) / 256, 256, 0, stream>>>(
                (const float*)(ws + o_xz), conv_w[p], conv_b[p],
                (float*)(ws + o_xc32), (h16*)(ws + o_xc16), L, convTotal, dconv[p], dir);
            gemm(o_xc16, o_wxpw[p], (float*)(ws + o_xdbl), nullptr, nullptr, nullptr,
                 XDBLN, 256, 0);
            scan_kernel<<<B, 256, 0, stream>>>(
                (const float*)(ws + o_xdbl), dt_w[p], dt_b[p], (const float*)(ws + o_xc32),
                (const float*)(ws + o_xz), A_log[p], Dp[p], (h16*)(ws + o_yg16), L, dir);
            float* fbuf = (float*)(ws + (dir == 0 ? o_f0 : o_f1));
            gemm(o_yg16, o_wopw[p], fbuf, nullptr, nullptr,
                 (p == 0) ? nullptr : fbuf, 128, 256, 0);    // accumulate ml+mg
        }
    }

    // ---- gate + fuse ----
    int nfeat = BL * DMODEL;
    cat_kernel<<<(nfeat + 255) / 256, 256, 0, stream>>>(
        (const float*)(ws + o_f0), (const float*)(ws + o_f1), (h16*)(ws + o_cat), nfeat);
    gemm(o_cat, o_wgate, (float*)(ws + o_tmp), nullptr, gate_b, nullptr, 128, 256, 0);
    fuse_kernel<<<(nfeat + 255) / 256, 256, 0, stream>>>(
        (const float*)(ws + o_tmp), (const float*)(ws + o_f0), (const float*)(ws + o_f1),
        (h16*)(ws + o_xn16), nfeat);

    // ---- bi output proj + residual: x2 = fused @ out_w^T + out_b + x ----
    gemm(o_xn16, o_woutw, (float*)(ws + o_x2), nullptr, out_b, x, 128, 128, 0);

    // ---- FFN: LN -> GELU(fc1) -> fc2 + residual ----
    ln128_kernel<<<BL, 128, 0, stream>>>((const float*)(ws + o_x2), fng, fnb,
                                         nullptr, (h16*)(ws + o_yg16));
    gemm(o_yg16, o_wfc1, nullptr, (h16*)(ws + o_xc16), fc1b, nullptr, 256, 128, 1);
    gemm(o_xc16, o_wfc2, (float*)d_out, nullptr, fc2b, (const float*)(ws + o_x2), 128, 256, 0);
}
